// DyHeadBlock_25374666785292
// MI455X (gfx1250) — matrix operand order repaired and checked by reading, not compiled, not run on hardware
//
#include <hip/hip_runtime.h>
#include <math.h>

typedef __attribute__((ext_vector_type(16))) _Float16 v16h;
typedef __attribute__((ext_vector_type(8)))  _Float16 v8h;
typedef __attribute__((ext_vector_type(8)))  float    v8f;

#define CIN 256
#define KDIM 2304           // CIN * 9
#define GN_G 16
#define GN_EPS 1e-5f

__device__ __forceinline__ float hsigf(float x){
    float t = x + 3.0f;
    t = fminf(fmaxf(t, 0.0f), 6.0f);
    return t * (1.0f/6.0f);
}

// ---------------- offset conv 3x3 (256 -> 27), sigmoid on mask channels ----
__global__ void conv_off_kernel(const float* __restrict__ x,
                                const float* __restrict__ w,
                                const float* __restrict__ b,
                                float* __restrict__ om,
                                int B, int H, int W){
    long long HW = (long long)H * W;
    long long idx = (long long)blockIdx.x * blockDim.x + threadIdx.x;
    long long total = (long long)B * 27 * HW;
    if (idx >= total) return;
    int pw = (int)(idx % W);
    int ph = (int)((idx / W) % H);
    int oc = (int)((idx / HW) % 27);
    int bb = (int)(idx / (27 * HW));
    float acc = b[oc];
    const float* xb = x + (size_t)bb * CIN * HW;
    const float* wo = w + (size_t)oc * CIN * 9;
    for (int ky = 0; ky < 3; ky++){
        int y = ph + ky - 1;
        if ((unsigned)y >= (unsigned)H) continue;
        for (int kx = 0; kx < 3; kx++){
            int xx = pw + kx - 1;
            if ((unsigned)xx >= (unsigned)W) continue;
            const float* xp = xb + (size_t)y * W + xx;
            const float* wp = wo + ky*3 + kx;
            for (int ci = 0; ci < CIN; ci++)
                acc = fmaf(xp[(size_t)ci * HW], wp[(size_t)ci * 9], acc);
        }
    }
    if (oc >= 18) acc = 1.0f / (1.0f + __expf(-acc));
    om[idx] = acc;
}

// ------------- DCNv2 bilinear gather: build val[M][2304] in f16 ------------
__global__ void sample_kernel(const float* __restrict__ x,
                              const float* __restrict__ om,
                              _Float16* __restrict__ val,
                              int B, int Hin, int Win, int Ho, int Wo,
                              int Hoff, int Woff, int stride){
    long long idx = (long long)blockIdx.x * blockDim.x + threadIdx.x;
    long long total = (long long)B * Ho * Wo * 9;
    if (idx >= total) return;
    int tap = (int)(idx % 9);
    long long m = idx / 9;
    int wo_ = (int)(m % Wo);
    int ho  = (int)((m / Wo) % Ho);
    int bb  = (int)(m / ((long long)Wo * Ho));

    size_t offHW = (size_t)Hoff * Woff;
    const float* omb = om + (size_t)bb * 27 * offHW + (size_t)ho * Woff + wo_;
    float oy  = omb[(size_t)(2*tap)   * offHW];
    float ox  = omb[(size_t)(2*tap+1) * offHW];
    float msk = omb[(size_t)(18+tap)  * offHW];

    float py = (float)(ho * stride - 1 + tap / 3) + oy;
    float px = (float)(wo_ * stride - 1 + tap % 3) + ox;
    float y0f = floorf(py), x0f = floorf(px);
    int iy0 = (int)y0f, ix0 = (int)x0f;
    float wy = py - y0f, wx = px - x0f;
    int iy1 = iy0 + 1, ix1 = ix0 + 1;
    float vy0 = (iy0 >= 0 && iy0 < Hin) ? 1.0f : 0.0f;
    float vy1 = (iy1 >= 0 && iy1 < Hin) ? 1.0f : 0.0f;
    float vx0 = (ix0 >= 0 && ix0 < Win) ? 1.0f : 0.0f;
    float vx1 = (ix1 >= 0 && ix1 < Win) ? 1.0f : 0.0f;
    int cy0 = min(max(iy0, 0), Hin-1), cy1 = min(max(iy1, 0), Hin-1);
    int cx0 = min(max(ix0, 0), Win-1), cx1 = min(max(ix1, 0), Win-1);
    float w00 = (1.0f-wy)*(1.0f-wx) * vy0*vx0 * msk;
    float w01 = (1.0f-wy)*wx        * vy0*vx1 * msk;
    float w10 = wy*(1.0f-wx)        * vy1*vx0 * msk;
    float w11 = wy*wx               * vy1*vx1 * msk;

    size_t chs = (size_t)Hin * Win;
    const float* base = x + (size_t)bb * CIN * chs;
    const float* p00 = base + (size_t)cy0 * Win + cx0;
    const float* p01 = base + (size_t)cy0 * Win + cx1;
    const float* p10 = base + (size_t)cy1 * Win + cx0;
    const float* p11 = base + (size_t)cy1 * Win + cx1;
    _Float16* vrow = val + (size_t)m * KDIM + tap;
    for (int c = 0; c < CIN; c++){
        float v = w00*p00[0] + w01*p01[0] + w10*p10[0] + w11*p11[0];
        vrow[(size_t)c * 9] = (_Float16)v;
        p00 += chs; p01 += chs; p10 += chs; p11 += chs;
    }
}

// ---------------- WMMA GEMM: feat[b][o][pix] = val[M][K] x wt[O][K]^T ------
// Per wave: 32(M) x 64(N) tile -> 8 accumulators; each B fragment feeds 2 WMMAs.
__global__ void __launch_bounds__(128)
mdcn_gemm_kernel(const _Float16* __restrict__ val,
                 const _Float16* __restrict__ wt,
                 const float* __restrict__ bias,
                 float* __restrict__ feat,
                 int M, int HW){
    int wave = threadIdx.x >> 5;
    int lane = threadIdx.x & 31;
    int ln = lane & 15;
    int hi = lane >> 4;
    int m0 = blockIdx.x * 32;
    int o0 = wave * 64;

    v8f acc[2][4] = {{{}, {}, {}, {}}, {{}, {}, {}, {}}};
    const _Float16* arow0 = val + (size_t)(m0 + ln) * KDIM;
    const _Float16* arow1 = arow0 + (size_t)16 * KDIM;
    const _Float16* brow[4];
#pragma unroll
    for (int nt = 0; nt < 4; nt++)
        brow[nt] = wt + (size_t)(o0 + nt*16 + ln) * KDIM;

    for (int k0 = 0; k0 < KDIM; k0 += 32){
        // A fragments: 16x32 f16, lane holds row (lane&15), K contiguous (k = 16*hi + s)
        v8h a0lo = *(const v8h*)(arow0 + k0 + hi*16);
        v8h a0hi = *(const v8h*)(arow0 + k0 + hi*16 + 8);
        v8h a1lo = *(const v8h*)(arow1 + k0 + hi*16);
        v8h a1hi = *(const v8h*)(arow1 + k0 + hi*16 + 8);
        v16h a0, a1;
#pragma unroll
        for (int i = 0; i < 8; i++){
            a0[i] = a0lo[i]; a0[8+i] = a0hi[i];
            a1[i] = a1lo[i]; a1[8+i] = a1hi[i];
        }
        int kb = k0 + hi*16;
#pragma unroll
        for (int nt = 0; nt < 4; nt++){
            // B fragment: 32x16 f16, lane = column, 16 contiguous K halves
            v16h bfr = *(const v16h*)(brow[nt] + kb);
            acc[0][nt] = __builtin_amdgcn_wmma_f32_16x16x32_f16(
                false, a0, false, bfr, (short)0, acc[0][nt], false, false);
            acc[1][nt] = __builtin_amdgcn_wmma_f32_16x16x32_f16(
                false, a1, false, bfr, (short)0, acc[1][nt], false, false);
        }
    }
#pragma unroll
    for (int mt = 0; mt < 2; mt++){
#pragma unroll
        for (int nt = 0; nt < 4; nt++){
            int ocol = o0 + nt*16 + ln;
            float bv = bias[ocol];
#pragma unroll
            for (int r = 0; r < 8; r++){
                int m = m0 + mt*16 + hi*8 + r;
                int bimg = m / HW;
                int pix = m - bimg * HW;
                feat[((size_t)bimg * CIN + ocol) * HW + pix] = acc[mt][nt][r] + bv;
            }
        }
    }
}

// ---------------- GroupNorm stats + apply -----------------------------------
__global__ void gn_stats_kernel(const float* __restrict__ feat,
                                float* __restrict__ stats, int HW){
    int bg = blockIdx.x;                      // b*16 + g
    size_t base = (size_t)bg * 16 * HW;       // (b*256 + g*16)*HW
    long long N = (long long)16 * HW;
    float s = 0.0f, ss = 0.0f;
    for (long long i = threadIdx.x; i < N; i += blockDim.x){
        float v = feat[base + i];
        s += v; ss += v*v;
    }
    __shared__ float sh[256], sh2[256];
    sh[threadIdx.x] = s; sh2[threadIdx.x] = ss;
    __syncthreads();
    for (int t = 128; t > 0; t >>= 1){
        if ((int)threadIdx.x < t){ sh[threadIdx.x] += sh[threadIdx.x+t]; sh2[threadIdx.x] += sh2[threadIdx.x+t]; }
        __syncthreads();
    }
    if (threadIdx.x == 0){
        float mu = sh[0] / (float)N;
        float var = sh2[0] / (float)N - mu*mu;
        stats[2*bg]   = mu;
        stats[2*bg+1] = rsqrtf(var + GN_EPS);
    }
}

__global__ void gn_apply_kernel(float* __restrict__ feat,
                                const float* __restrict__ stats,
                                const float* __restrict__ gamma,
                                const float* __restrict__ beta,
                                long long total, int HW){
    long long idx = (long long)blockIdx.x * blockDim.x + threadIdx.x;
    if (idx >= total) return;
    int c = (int)((idx / HW) % CIN);
    int b = (int)(idx / ((long long)CIN * HW));
    int bg = b * 16 + (c >> 4);
    feat[idx] = (feat[idx] - stats[2*bg]) * stats[2*bg+1] * gamma[c] + beta[c];
}

// ---------------- per-(b,c) spatial mean pooling ----------------------------
__global__ void pool_kernel(const float* __restrict__ feat,
                            float* __restrict__ pooled, int HW, float scale){
    int bc = blockIdx.x;
    size_t base = (size_t)bc * HW;
    float s = 0.0f;
    for (int i = threadIdx.x; i < HW; i += blockDim.x) s += feat[base + i];
    __shared__ float sh[256];
    sh[threadIdx.x] = s;
    __syncthreads();
    for (int t = 128; t > 0; t >>= 1){
        if ((int)threadIdx.x < t) sh[threadIdx.x] += sh[threadIdx.x+t];
        __syncthreads();
    }
    if (threadIdx.x == 0) pooled[bc] = sh[0] * scale / (float)HW;
}

// ---------------- scale attention: s[b] = hsig(relu(pooled . w + b)) --------
__global__ void attn_kernel(const float* __restrict__ pooled,
                            const float* __restrict__ w_attn,
                            const float* __restrict__ b_attn,
                            float* __restrict__ s){
    int b = blockIdx.x;
    float v = pooled[b*CIN + threadIdx.x] * w_attn[threadIdx.x];
    __shared__ float sh[256];
    sh[threadIdx.x] = v;
    __syncthreads();
    for (int t = 128; t > 0; t >>= 1){
        if ((int)threadIdx.x < t) sh[threadIdx.x] += sh[threadIdx.x+t];
        __syncthreads();
    }
    if (threadIdx.x == 0) s[b] = hsigf(fmaxf(sh[0] + b_attn[0], 0.0f));
}

// ---------------- sum_feat (set or add) feat * s[b] -------------------------
__global__ void accum_kernel(float* __restrict__ sum,
                             const float* __restrict__ feat,
                             const float* __restrict__ s,
                             long long total, int HW, int add){
    long long idx = (long long)blockIdx.x * blockDim.x + threadIdx.x;
    if (idx >= total) return;
    int b = (int)(idx / ((long long)CIN * HW));
    float v = feat[idx] * s[b];
    sum[idx] = add ? (sum[idx] + v) : v;
}

// ---------------- bilinear resize (align-corners linspace) ------------------
__global__ void resize_kernel(const float* __restrict__ in, float* __restrict__ out,
                              int Hs, int Ws, int H, int W, long long total){
    long long idx = (long long)blockIdx.x * blockDim.x + threadIdx.x;
    if (idx >= total) return;
    int w = (int)(idx % W);
    int h = (int)((idx / W) % H);
    long long bc = idx / ((long long)H * W);
    float ys = (H > 1) ? (float)h * (float)(Hs - 1) / (float)(H - 1) : 0.0f;
    float xs = (W > 1) ? (float)w * (float)(Ws - 1) / (float)(W - 1) : 0.0f;
    int y0 = min(max((int)floorf(ys), 0), Hs - 2);
    int x0 = min(max((int)floorf(xs), 0), Ws - 2);
    float wy = ys - (float)y0, wx = xs - (float)x0;
    const float* p = in + bc * (size_t)Hs * Ws;
    float r0 = p[(size_t)y0*Ws + x0]*(1.0f-wx)     + p[(size_t)y0*Ws + x0 + 1]*wx;
    float r1 = p[(size_t)(y0+1)*Ws + x0]*(1.0f-wx) + p[(size_t)(y0+1)*Ws + x0 + 1]*wx;
    out[idx] = r0*(1.0f-wy) + r1*wy;
}

// ---------------- DyReLU coefficient MLP ------------------------------------
__global__ void dyrelu_mlp_kernel(const float* __restrict__ pooled,
                                  const float* __restrict__ w1, const float* __restrict__ b1,
                                  const float* __restrict__ w2, const float* __restrict__ b2,
                                  float* __restrict__ dyc){
    int b = blockIdx.x;
    int t = threadIdx.x;               // 256 threads
    __shared__ float hid[64];
    if (t < 64){
        float a = b1[t];
        const float* wr = w1 + (size_t)t * CIN;
        const float* pp = pooled + (size_t)b * CIN;
        for (int c = 0; c < CIN; c++) a = fmaf(pp[c], wr[c], a);
        hid[t] = fmaxf(a, 0.0f);
    }
    __syncthreads();
    for (int q = 0; q < 4; q++){
        int j = q * CIN + t;
        float a = b2[j];
        const float* wr = w2 + (size_t)j * 64;
        for (int i = 0; i < 64; i++) a = fmaf(hid[i], wr[i], a);
        float v = hsigf(a);
        float o;
        if (q == 0)      o = (v - 0.5f) * 2.0f + 1.0f;
        else if (q == 2) o = (v - 0.5f) * 2.0f;
        else             o = v - 0.5f;
        dyc[((size_t)b * CIN + t) * 4 + q] = o;
    }
}

__global__ void dyrelu_apply_kernel(const float* __restrict__ sum,
                                    const float* __restrict__ dyc,
                                    float* __restrict__ out,
                                    long long total, int HW, float invn){
    long long idx = (long long)blockIdx.x * blockDim.x + threadIdx.x;
    if (idx >= total) return;
    int c = (int)((idx / HW) % CIN);
    int b = (int)(idx / ((long long)CIN * HW));
    float x = sum[idx] * invn;
    const float* d = dyc + ((size_t)b * CIN + c) * 4;
    out[idx] = fmaxf(x*d[0] + d[1], x*d[2] + d[3]);
}

// ---------------- f32 -> f16 weight conversion ------------------------------
__global__ void cvt_h_kernel(const float* __restrict__ src, _Float16* __restrict__ dst, int n){
    int i = blockIdx.x * blockDim.x + threadIdx.x;
    if (i < n) dst[i] = (_Float16)src[i];
}

// ============================================================================
extern "C" void kernel_launch(void* const* d_in, const int* in_sizes, int n_in,
                              void* d_out, int out_size, void* d_ws, size_t ws_size,
                              hipStream_t stream) {
    (void)in_sizes; (void)n_in; (void)out_size; (void)ws_size;
    const float* x[3]   = {(const float*)d_in[0], (const float*)d_in[1], (const float*)d_in[2]};
    const float* w_off  = (const float*)d_in[3];
    const float* b_off  = (const float*)d_in[4];
    const float* w_mid  = (const float*)d_in[5];
    const float* b_mid  = (const float*)d_in[6];
    const float* gm_mid = (const float*)d_in[7];
    const float* gb_mid = (const float*)d_in[8];
    const float* w_low  = (const float*)d_in[9];
    const float* b_low  = (const float*)d_in[10];
    const float* gm_low = (const float*)d_in[11];
    const float* gb_low = (const float*)d_in[12];
    const float* w_high = (const float*)d_in[13];
    const float* b_high = (const float*)d_in[14];
    const float* gm_high= (const float*)d_in[15];
    const float* gb_high= (const float*)d_in[16];
    const float* w_attn = (const float*)d_in[17];
    const float* b_attn = (const float*)d_in[18];
    const float* w1p    = (const float*)d_in[19];
    const float* b1p    = (const float*)d_in[20];
    const float* w2p    = (const float*)d_in[21];
    const float* b2p    = (const float*)d_in[22];

    const int B = 2;
    const int Hs[3] = {64, 32, 16};

    // ---- carve workspace ----
    char* ws = (char*)d_ws;
    size_t off = 0;
    auto take = [&](size_t bytes) -> char* {
        char* p = ws + off;
        off = (off + bytes + 255) & ~(size_t)255;
        return p;
    };
    const int WN = CIN * KDIM;                               // 589,824 weights each
    _Float16* wtm   = (_Float16*)take((size_t)WN * 2);
    _Float16* wtl   = (_Float16*)take((size_t)WN * 2);
    _Float16* wth   = (_Float16*)take((size_t)WN * 2);
    float*    om    = (float*)take((size_t)B * 27 * 64 * 64 * 4);
    _Float16* val   = (_Float16*)take((size_t)B * 64 * 64 * KDIM * 2);   // 37.7 MB max
    float*    feat  = (float*)take((size_t)B * CIN * 64 * 64 * 4);
    float*    rsz   = (float*)take((size_t)B * CIN * 64 * 64 * 4);
    float*    sumf  = (float*)take((size_t)B * CIN * 64 * 64 * 4);
    float*    stats = (float*)take((size_t)B * GN_G * 2 * 4);
    float*    pooled= (float*)take((size_t)B * CIN * 4);
    float*    sbuf  = (float*)take(256);
    float*    dyc   = (float*)take((size_t)B * CIN * 4 * 4);

    // ---- convert DCN weights to f16 (layout already [O][Cin*9] = [O][K]) ----
    cvt_h_kernel<<<(WN + 255) / 256, 256, 0, stream>>>(w_mid,  wtm, WN);
    cvt_h_kernel<<<(WN + 255) / 256, 256, 0, stream>>>(w_low,  wtl, WN);
    cvt_h_kernel<<<(WN + 255) / 256, 256, 0, stream>>>(w_high, wth, WN);

    auto run_branch = [&](const float* xin, int Hin, int Ho, int stride,
                          const _Float16* wt, const float* bias,
                          const float* gm, const float* gb, int Hoff){
        int Win = Hin, Wo = Ho, Woff = Hoff;
        int M = B * Ho * Wo;
        int HW = Ho * Wo;
        long long stot = (long long)M * 9;
        sample_kernel<<<(int)((stot + 255) / 256), 256, 0, stream>>>(
            xin, om, val, B, Hin, Win, Ho, Wo, Hoff, Woff, stride);
        mdcn_gemm_kernel<<<M / 32, 128, 0, stream>>>(val, wt, bias, feat, M, HW);
        gn_stats_kernel<<<B * GN_G, 256, 0, stream>>>(feat, stats, HW);
        long long ftot = (long long)B * CIN * HW;
        gn_apply_kernel<<<(int)((ftot + 255) / 256), 256, 0, stream>>>(
            feat, stats, gm, gb, ftot, HW);
    };
    auto attn_accum = [&](const float* f, int HW, int add){
        pool_kernel<<<B * CIN, 256, 0, stream>>>(f, pooled, HW, 1.0f);
        attn_kernel<<<B, 256, 0, stream>>>(pooled, w_attn, b_attn, sbuf);
        long long tot = (long long)B * CIN * HW;
        accum_kernel<<<(int)((tot + 255) / 256), 256, 0, stream>>>(sumf, f, sbuf, tot, HW, add);
    };

    long long out_off = 0;
    for (int l = 0; l < 3; l++){
        int H = Hs[l], W = H, HW = H * W;
        long long ctot = (long long)B * 27 * HW;
        conv_off_kernel<<<(int)((ctot + 255) / 256), 256, 0, stream>>>(
            x[l], w_off, b_off, om, B, H, W);

        // mid branch (stride 1, same level)
        run_branch(x[l], H, H, 1, wtm, b_mid, gm_mid, gb_mid, H);
        attn_accum(feat, HW, 0);
        int n = 1;

        // low branch (stride 2 from finer level)
        if (l > 0){
            run_branch(x[l-1], Hs[l-1], H, 2, wtl, b_low, gm_low, gb_low, H);
            attn_accum(feat, HW, 1);
            n++;
        }

        // high branch (stride 1 on coarser level, offsets cropped, then resize up)
        if (l < 2){
            int Hh = Hs[l+1];
            run_branch(x[l+1], Hh, Hh, 1, wth, b_high, gm_high, gb_high, H);
            long long rtot = (long long)B * CIN * HW;
            resize_kernel<<<(int)((rtot + 255) / 256), 256, 0, stream>>>(
                feat, rsz, Hh, Hh, H, W, rtot);
            attn_accum(rsz, HW, 1);
            n++;
        }

        // DyReLU on sum/n
        float invn = 1.0f / (float)n;
        pool_kernel<<<B * CIN, 256, 0, stream>>>(sumf, pooled, HW, invn);
        dyrelu_mlp_kernel<<<B, 256, 0, stream>>>(pooled, w1p, b1p, w2p, b2p, dyc);
        long long ftot = (long long)B * CIN * HW;
        dyrelu_apply_kernel<<<(int)((ftot + 255) / 256), 256, 0, stream>>>(
            sumf, dyc, (float*)d_out + out_off, ftot, HW, invn);
        out_off += ftot;
    }
}
